// CrossNetMix_70592082477156
// MI455X (gfx1250) — matrix operand order repaired and checked by reading, not compiled, not run on hardware
//
#include <hip/hip_runtime.h>

// CrossNetMix for MI455X (gfx1250): 3 layers x (GEMM1 [B,288]=[B,2048]x[2048,288] w/ gate cols,
// per-expert 64x64 mix GEMM, GEMM3 [B,2048]=[B,256]x[256,2048] + fused residual epilogue),
// all matrix math on v_wmma_f32_16x16x32_bf16, fp32 accumulate.

#define B_  8192
#define D_  2048
#define R_  64
#define E_  4
#define L_  3
#define N1_ 288   // 256 (E*R) + 4 gate cols, padded to 18*16

typedef __attribute__((ext_vector_type(16))) __bf16 v16bf;
typedef __attribute__((ext_vector_type(8)))  float  v8f;
typedef __attribute__((ext_vector_type(8)))  unsigned short u16x8;
typedef __attribute__((ext_vector_type(4)))  unsigned short u16x4;
typedef __attribute__((ext_vector_type(4)))  float  f32x4;

union BFrag { u16x8 h[2]; v16bf v; };

__device__ __forceinline__ unsigned short f2bf(float f) {
  unsigned int u = __float_as_uint(f);
  u += 0x7fffu + ((u >> 16) & 1u);           // round-to-nearest-even
  return (unsigned short)(u >> 16);
}

__device__ __forceinline__ v8f wmma_bf16(const BFrag& a, const BFrag& b, v8f c) {
  // (neg_a, A, neg_b, B, c_mod, C, reuse_a, reuse_b)
  return __builtin_amdgcn_wmma_f32_16x16x32_bf16(false, a.v, false, b.v, (short)0, c, false, false);
}

// A-fragment (16x32 bf16): lane m = lane&15, half = lane>>4.
// elem s -> K=16*half+s
// => two contiguous 8-ushort loads at K-offsets 16*half and 16*half+8.
__device__ __forceinline__ void load_afrag(BFrag& a, const unsigned short* row, int half) {
  a.h[0] = *(const u16x8*)(row + 16 * half);
  a.h[1] = *(const u16x8*)(row + 16 * half + 8);
}
// B-fragment (32x16 bf16), B stored column-major in LDS (Bs[n][k]):
// lane n = lane&15, half = lane>>4 holds K = 16*half .. 16*half+15.
__device__ __forceinline__ void load_bfrag(BFrag& b, const unsigned short* col, int half) {
  b.h[0] = *(const u16x8*)(col + 16 * half);
  b.h[1] = *(const u16x8*)(col + 16 * half + 8);
}

// ---------------------------------------------------------------------------
// Weight packing (runs every launch; deterministic).
// Wv[l][k=0..2047][n=0..287]: n<256 -> V[l, e=n>>6, k, r=n&63]; 256<=n<260 -> gate_w[n-256][k]; else 0
// Wu[l][k=0..255][d]:          U[l, e=k>>6, d, s=k&63]
// Wc[l][n=e*64+s][r]:          C[l, e, s, r]
// ---------------------------------------------------------------------------
__global__ __launch_bounds__(256) void pack_weights(
    const float* __restrict__ U, const float* __restrict__ V,
    const float* __restrict__ C, const float* __restrict__ gw,
    unsigned short* __restrict__ Wv, unsigned short* __restrict__ Wu,
    unsigned short* __restrict__ Wc)
{
  const int NV = L_ * D_ * N1_;
  const int NU = L_ * 256 * D_;
  const int NC = L_ * 256 * R_;
  int idx = blockIdx.x * 256 + threadIdx.x;
  if (idx < NV) {
    int l = idx / (D_ * N1_); int rem = idx - l * (D_ * N1_);
    int k = rem / N1_;        int n   = rem - k * N1_;
    float v = 0.0f;
    if (n < 256)      v = V[(((size_t)l * E_ + (n >> 6)) * D_ + k) * R_ + (n & 63)];
    else if (n < 260) v = gw[(size_t)(n - 256) * D_ + k];
    Wv[idx] = f2bf(v);
  } else if (idx < NV + NU) {
    int j = idx - NV;
    int l = j / (256 * D_); int rem = j - l * (256 * D_);
    int k = rem / D_;       int d   = rem - k * D_;
    Wu[j] = f2bf(U[(((size_t)l * E_ + (k >> 6)) * D_ + d) * R_ + (k & 63)]);
  } else if (idx < NV + NU + NC) {
    int j = idx - NV - NU;
    int l = j / (256 * R_); int rem = j - l * (256 * R_);
    int n = rem / R_;       int r   = rem - n * R_;
    int e = n >> 6, s = n & 63;
    Wc[j] = f2bf(C[(((size_t)l * E_ + e) * R_ + s) * R_ + r]);
  }
}

// ---------------------------------------------------------------------------
// GEMM1: VXout[b, 0..255] = tanh(x . Vcat), GL[b, 0..3] = gate logits.
// 128 thr = 4 waves; M-tile 32 (2 row groups x 16), N = 288 (each wave 9 tiles).
// ---------------------------------------------------------------------------
__global__ __launch_bounds__(128) void layer_gemm1(
    const float* __restrict__ X, const unsigned short* __restrict__ Wv,
    unsigned short* __restrict__ VX, float* __restrict__ GL)
{
  __shared__ unsigned short As[32][40];     // 32 rows x 32 K, pad->40
  __shared__ unsigned short Bs[N1_][40];    // col-major: [n][k]
  const int tid  = threadIdx.x;
  const int wave = tid >> 5, lane = tid & 31;
  const int hf   = lane >> 4, lc = lane & 15;
  const int m0   = blockIdx.x * 32;
  const int mw   = (wave >> 1) * 16;
  const int nb   = (wave & 1) * 9;          // first n-tile of this wave

  v8f acc[9];
  const v8f z = {0.f,0.f,0.f,0.f,0.f,0.f,0.f,0.f};
#pragma unroll
  for (int t = 0; t < 9; ++t) acc[t] = z;

  for (int k0 = 0; k0 < D_; k0 += 32) {
    __syncthreads();
    { // A: 32x32, f32 -> bf16
      int r = tid >> 2, c = (tid & 3) * 8;
      const float* src = X + (size_t)(m0 + r) * D_ + k0 + c;
      f32x4 f0 = *(const f32x4*)src;
      f32x4 f1 = *(const f32x4*)(src + 4);
      unsigned short* dst = &As[r][c];
      dst[0]=f2bf(f0[0]); dst[1]=f2bf(f0[1]); dst[2]=f2bf(f0[2]); dst[3]=f2bf(f0[3]);
      dst[4]=f2bf(f1[0]); dst[5]=f2bf(f1[1]); dst[6]=f2bf(f1[2]); dst[7]=f2bf(f1[3]);
    }
    // B: 32 x 288 bf16, transpose into Bs[n][k]
#pragma unroll 2
    for (int q = tid; q < 32 * (N1_ / 4); q += 128) {
      int k = q / (N1_ / 4);
      int n = (q - k * (N1_ / 4)) * 4;
      u16x4 w = *(const u16x4*)(Wv + (size_t)(k0 + k) * N1_ + n);
      Bs[n+0][k] = w[0]; Bs[n+1][k] = w[1]; Bs[n+2][k] = w[2]; Bs[n+3][k] = w[3];
    }
    if (k0 + 32 < D_) __builtin_prefetch(Wv + (size_t)(k0 + 32) * N1_, 0, 1);
    __syncthreads();

    BFrag a;
    load_afrag(a, &As[mw + lc][0], hf);
#pragma unroll
    for (int t = 0; t < 9; ++t) {
      BFrag b;
      load_bfrag(b, &Bs[(nb + t) * 16 + lc][0], hf);
      acc[t] = wmma_bf16(a, b, acc[t]);
    }
  }

  // epilogue: cols<256 -> tanh->bf16 VX; cols 256..259 -> raw gate logits
#pragma unroll
  for (int t = 0; t < 9; ++t) {
    int n = (nb + t) * 16 + lc;
#pragma unroll
    for (int i = 0; i < 8; ++i) {
      int b = m0 + mw + i + 8 * hf;
      float v = acc[t][i];
      if (n < 256)       VX[(size_t)b * 256 + n] = f2bf(tanhf(v));
      else if (n < 260)  GL[(size_t)b * 4 + (n - 256)] = v;
    }
  }
}

// ---------------------------------------------------------------------------
// GEMM2 (per-expert 64x64 mix): W[b, e*64+s] = softmax(GL)[b,e]*tanh(sum_r Wc[n][r]*vx[b,e*64+r])
// M-tile 32, N = 256; wave covers 2 experts (8 n-tiles), K=64 per expert.
// ---------------------------------------------------------------------------
__global__ __launch_bounds__(128) void layer_gemm2(
    const unsigned short* __restrict__ VX, const unsigned short* __restrict__ Wc,
    const float* __restrict__ GL, unsigned short* __restrict__ Wb)
{
  __shared__ unsigned short Vs[32][264];    // A: 32 x 256, pad->264
  __shared__ unsigned short Cs[256][72];    // B col-major: [n=e*64+s][k=r], pad->72
  const int tid  = threadIdx.x;
  const int wave = tid >> 5, lane = tid & 31;
  const int hf   = lane >> 4, lc = lane & 15;
  const int m0   = blockIdx.x * 32;
  const int mw   = (wave >> 1) * 16;
  const int e0   = (wave & 1) * 2;

#pragma unroll 4
  for (int q = tid; q < 2048; q += 128) {       // 32*256/4 quads
    int r = q >> 6, c = (q & 63) * 4;
    *(u16x4*)&Vs[r][c] = *(const u16x4*)(VX + (size_t)(m0 + r) * 256 + c);
  }
#pragma unroll 4
  for (int q = tid; q < 4096; q += 128) {       // 256*64/4 quads (already [n][r])
    int n = q >> 4, c = (q & 15) * 4;
    *(u16x4*)&Cs[n][c] = *(const u16x4*)(Wc + (size_t)n * R_ + c);
  }
  __syncthreads();

  v8f acc[8];
  const v8f z = {0.f,0.f,0.f,0.f,0.f,0.f,0.f,0.f};
#pragma unroll
  for (int t = 0; t < 8; ++t) acc[t] = z;

#pragma unroll
  for (int ei = 0; ei < 2; ++ei) {
    int e = e0 + ei;
#pragma unroll
    for (int ks = 0; ks < 2; ++ks) {
      BFrag a;
      load_afrag(a, &Vs[mw + lc][e * 64 + ks * 32], hf);
#pragma unroll
      for (int q = 0; q < 4; ++q) {
        BFrag b;
        load_bfrag(b, &Cs[e * 64 + q * 16 + lc][ks * 32], hf);
        acc[ei * 4 + q] = wmma_bf16(a, b, acc[ei * 4 + q]);
      }
    }
  }

#pragma unroll
  for (int i = 0; i < 8; ++i) {
    int b = m0 + mw + i + 8 * hf;
    f32x4 g = *(const f32x4*)(GL + (size_t)b * 4);
    float mx = fmaxf(fmaxf(g[0], g[1]), fmaxf(g[2], g[3]));
    float ex0 = __expf(g[0]-mx), ex1 = __expf(g[1]-mx), ex2 = __expf(g[2]-mx), ex3 = __expf(g[3]-mx);
    float inv = 1.0f / (ex0 + ex1 + ex2 + ex3);
    float gs[4] = { ex0*inv, ex1*inv, ex2*inv, ex3*inv };
#pragma unroll
    for (int ei = 0; ei < 2; ++ei) {
#pragma unroll
      for (int q = 0; q < 4; ++q) {
        float w = gs[e0 + ei] * tanhf(acc[ei * 4 + q][i]);
        Wb[(size_t)b * 256 + (e0 + ei) * 64 + q * 16 + lc] = f2bf(w);
      }
    }
  }
}

// ---------------------------------------------------------------------------
// GEMM3: Y = Wb(B,256) x Wu(256,2048); Xn = Xl + X0 * (bias + Y)
// M-tile 32, N-tile 64; K=256 resident in LDS (8 k-steps).
// ---------------------------------------------------------------------------
__global__ __launch_bounds__(128) void layer_gemm3(
    const unsigned short* __restrict__ Wb, const unsigned short* __restrict__ Wu,
    const float* __restrict__ Xl, const float* __restrict__ X0,
    const float* __restrict__ bias, float* __restrict__ Xn)
{
  __shared__ unsigned short Asd[32][264];   // A: 32 x 256
  __shared__ unsigned short Bsd[64][264];   // B col-major: [n][k=256]
  const int tid  = threadIdx.x;
  const int wave = tid >> 5, lane = tid & 31;
  const int hf   = lane >> 4, lc = lane & 15;
  const int n0   = blockIdx.x * 64;
  const int m0   = blockIdx.y * 32;
  const int mw   = (wave >> 1) * 16;
  const int nwo  = (wave & 1) * 32;

#pragma unroll 4
  for (int q = tid; q < 2048; q += 128) {       // A quads
    int r = q >> 6, c = (q & 63) * 4;
    *(u16x4*)&Asd[r][c] = *(const u16x4*)(Wb + (size_t)(m0 + r) * 256 + c);
  }
#pragma unroll 4
  for (int q = tid; q < 4096; q += 128) {       // B: 256 x 64, transpose
    int k = q >> 4, c = (q & 15) * 4;
    u16x4 w = *(const u16x4*)(Wu + (size_t)k * D_ + n0 + c);
    Bsd[c+0][k] = w[0]; Bsd[c+1][k] = w[1]; Bsd[c+2][k] = w[2]; Bsd[c+3][k] = w[3];
  }
  __syncthreads();

  v8f acc[2];
  const v8f z = {0.f,0.f,0.f,0.f,0.f,0.f,0.f,0.f};
  acc[0] = z; acc[1] = z;

#pragma unroll
  for (int ks = 0; ks < 8; ++ks) {
    BFrag a;
    load_afrag(a, &Asd[mw + lc][ks * 32], hf);
#pragma unroll
    for (int t = 0; t < 2; ++t) {
      BFrag b;
      load_bfrag(b, &Bsd[nwo + t * 16 + lc][ks * 32], hf);
      acc[t] = wmma_bf16(a, b, acc[t]);
    }
  }

#pragma unroll
  for (int t = 0; t < 2; ++t) {
    int d = n0 + nwo + t * 16 + lc;
    float bi = bias[d];
#pragma unroll
    for (int i = 0; i < 8; ++i) {
      int b = m0 + mw + i + 8 * hf;
      size_t o = (size_t)b * D_ + d;
      Xn[o] = Xl[o] + X0[o] * (bi + acc[t][i]);
    }
  }
}

// ---------------------------------------------------------------------------
extern "C" void kernel_launch(void* const* d_in, const int* in_sizes, int n_in,
                              void* d_out, int out_size, void* d_ws, size_t ws_size,
                              hipStream_t stream) {
  const float* inp  = (const float*)d_in[0];
  const float* U    = (const float*)d_in[1];
  const float* V    = (const float*)d_in[2];
  const float* C    = (const float*)d_in[3];
  const float* gw   = (const float*)d_in[4];
  const float* bias = (const float*)d_in[5];
  float* out = (float*)d_out;

  char* ws = (char*)d_ws;
  size_t off = 0;
  auto take = [&](size_t bytes) -> char* {
    off = (off + 255) & ~(size_t)255;
    char* p = ws + off;
    off += bytes;
    return p;
  };
  float*          Xa = (float*)         take((size_t)B_ * D_ * 4);
  unsigned short* VX = (unsigned short*)take((size_t)B_ * 256 * 2);
  float*          GL = (float*)         take((size_t)B_ * E_ * 4);
  unsigned short* Wb = (unsigned short*)take((size_t)B_ * 256 * 2);
  unsigned short* Wv = (unsigned short*)take((size_t)L_ * D_ * N1_ * 2);
  unsigned short* Wu = (unsigned short*)take((size_t)L_ * 256 * D_ * 2);
  unsigned short* Wc = (unsigned short*)take((size_t)L_ * 256 * R_ * 2);

  const int npack = L_ * D_ * N1_ + L_ * 256 * D_ + L_ * 256 * R_;
  pack_weights<<<(npack + 255) / 256, 256, 0, stream>>>(U, V, C, gw, Wv, Wu, Wc);

  const float* xin[3]  = { inp, out, Xa };
  float*       xout[3] = { out, Xa, out };
  for (int l = 0; l < L_; ++l) {
    layer_gemm1<<<B_ / 32, 128, 0, stream>>>(
        xin[l], Wv + (size_t)l * D_ * N1_, VX, GL);
    layer_gemm2<<<B_ / 32, 128, 0, stream>>>(
        VX, Wc + (size_t)l * 256 * R_, GL, Wb);
    layer_gemm3<<<dim3(D_ / 64, B_ / 32), 128, 0, stream>>>(
        Wb, Wu + (size_t)l * 256 * D_, xin[l], inp, bias + (size_t)l * D_, xout[l]);
  }
}
